// AttentionHead_76321568850503
// MI455X (gfx1250) — matrix operand order repaired and checked by reading, not compiled, not run on hardware
//
#include <hip/hip_runtime.h>
#include <stdint.h>

// Problem constants (from reference): B=4, S=4096, H=1024, D=64
#define BATCH 4
#define SEQ   4096
#define HID   1024
#define DH    64
#define LOG2E 1.4426950408889634f

typedef __attribute__((ext_vector_type(16))) __bf16 v16bf;
typedef __attribute__((ext_vector_type(2)))  __bf16 v2bf;
typedef __attribute__((ext_vector_type(8)))  float  v8f;

union ABReg {                 // one WMMA A/B operand: 16 bf16 per lane = 8 VGPRs
    v16bf v;
    uint4 q[2];
    unsigned int u[8];
};

// Native f32->bf16: pair of fptruncs -> v_cvt_pk_bf16_f32 (one VALU op)
__device__ __forceinline__ unsigned int pack2bf(float lo, float hi) {
    v2bf t;
    t.x = (__bf16)lo;
    t.y = (__bf16)hi;
    return __builtin_bit_cast(unsigned int, t);
}
__device__ __forceinline__ unsigned short f2bf1(float f) {
    __bf16 h = (__bf16)f;
    return __builtin_bit_cast(unsigned short, h);
}
__device__ __forceinline__ v8f wmma_bf16(v16bf a, v16bf b, v8f c) {
    // D(f32 16x16) = A(bf16 16x32) * B(bf16 32x16) + C
    return __builtin_amdgcn_wmma_f32_16x16x32_bf16(false, a, false, b, (short)0, c,
                                                   false, false);
}

// ---------------------------------------------------------------- bias -----
__global__ void bias_kernel(const int* __restrict__ mask, float* __restrict__ bias) {
    int i = blockIdx.x * blockDim.x + threadIdx.x;
    if (i < BATCH * SEQ) bias[i] = (mask[i] == 0) ? -1.0e12f : 0.0f;
}

// ------------------------------------------------- weight pre-conversion ---
// Wb bf16, laid out [3][64][1024] row-major (same as W): B-tiles become two
// contiguous b128 loads with zero conversion work in the GEMM hot loop.
__global__ void wprep_kernel(const float* __restrict__ Wk, const float* __restrict__ Wq,
                             const float* __restrict__ Wv, unsigned int* __restrict__ wb)
{
    const int PER = DH * HID / 2;                       // 32768 u32 per matrix
    int tid = blockIdx.x * blockDim.x + threadIdx.x;    // 0..98303
    if (tid >= 3 * PER) return;
    int m   = tid / PER;
    int off = (tid - m * PER) * 2;
    const float* W = (m == 0) ? Wk : (m == 1) ? Wq : Wv;
    float2 f = *(const float2*)(W + off);
    wb[tid] = pack2bf(f.x, f.y);
}

// ------------------------------------------- fused K/Q/V projection GEMM ---
// One wave per 16-row tile computes all three projections: x A-tile is read
// and converted ONCE and feeds 12 WMMAs (3 matrices x 4 n-tiles).
// All 12 B-tiles stage in DISTINCT registers so the 24 b128 loads issue as a
// clause and drain with progressive s_wait_loadcnt in front of each WMMA.
// K/Q stored [S][64] bf16; V stored transposed Vt[64][S] bf16.
// K is pre-scaled by 1/sqrt(D) so the attention kernel skips the score scale.
__global__ __launch_bounds__(128) void proj_kernel(
    const float* __restrict__ x, const unsigned short* __restrict__ wb,
    unsigned short* __restrict__ kb, unsigned short* __restrict__ qb,
    unsigned short* __restrict__ vt)
{
    const int lane = threadIdx.x & 31;
    const int half = lane >> 4;
    const int ln   = lane & 15;
    const int wid  = blockIdx.x * 4 + (threadIdx.x >> 5);   // 0..1023
    const int b    = wid >> 8;
    const int rowbase = (wid & 255) << 4;

    const float* xrow = x + (size_t)(b * SEQ + rowbase + ln) * HID;

    v8f acc[12];
    #pragma unroll
    for (int n = 0; n < 12; ++n)
        #pragma unroll
        for (int r = 0; r < 8; ++r) acc[n][r] = 0.0f;

    #pragma unroll 1
    for (int kh = 0; kh < HID; kh += 32) {
        // ---- issue ALL B-tile loads first (12 tiles, distinct registers)
        ABReg bm[12];
        #pragma unroll
        for (int t = 0; t < 12; ++t) {
            const int m  = t >> 2;
            const int nt = t & 3;
            const uint4* wr = (const uint4*)(wb
                + ((size_t)m * DH + nt * 16 + ln) * HID + kh + 16 * half);
            bm[t].q[0] = wr[0];
            bm[t].q[1] = wr[1];
        }

        // ---- A tile: x rows; elems 0..15 -> k=16*half+e
        ABReg a;
        {
            const float4* p0 = (const float4*)(xrow + kh + 16 * half);
            const float4* p1 = (const float4*)(xrow + kh + 16 * half + 8);
            float4 f0 = p0[0], f1 = p0[1], f2 = p1[0], f3 = p1[1];
            a.u[0] = pack2bf(f0.x, f0.y); a.u[1] = pack2bf(f0.z, f0.w);
            a.u[2] = pack2bf(f1.x, f1.y); a.u[3] = pack2bf(f1.z, f1.w);
            a.u[4] = pack2bf(f2.x, f2.y); a.u[5] = pack2bf(f2.z, f2.w);
            a.u[6] = pack2bf(f3.x, f3.y); a.u[7] = pack2bf(f3.z, f3.w);
        }

        // ---- drain loads into 12 WMMAs
        #pragma unroll
        for (int t = 0; t < 12; ++t)
            acc[t] = wmma_bf16(a.v, bm[t].v, acc[t]);
    }

    // C/D layout: elem r -> row rowbase + r + 8*half, lane -> col nt*16 + ln
    #pragma unroll
    for (int nt = 0; nt < 4; ++nt)
        #pragma unroll
        for (int r = 0; r < 8; ++r) {
            int srow = rowbase + r + 8 * half;
            int d    = nt * 16 + ln;
            kb[(size_t)(b * SEQ + srow) * DH + d] = f2bf1(acc[nt][r] * 0.125f);
            qb[(size_t)(b * SEQ + srow) * DH + d] = f2bf1(acc[4 + nt][r]);
            vt[(size_t)(b * DH + d) * SEQ + srow] = f2bf1(acc[8 + nt][r]);
        }
}

// --------------------------------------------------- flash attention core --
// Per wave: 16 output rows i. St[j][i] = Q_j . K_i  (A = Q rows, B = K^T),
// softmax over j lives on the register-element axis + one shfl_xor(16).
// O^T = V^T * P^T accumulated as 4 C tiles (d groups), normalized by 1/l.
// All loads of a j-block (Q, V, bias) are issued up-front so their latency
// overlaps the score WMMAs and the whole softmax section.
__global__ __launch_bounds__(128) void attn_kernel(
    const unsigned short* __restrict__ kb, const unsigned short* __restrict__ qb,
    const unsigned short* __restrict__ vt, const float* __restrict__ bias,
    float* __restrict__ out)
{
    const int lane = threadIdx.x & 31;
    const int half = lane >> 4;
    const int ln   = lane & 15;
    const int wid  = blockIdx.x * 4 + (threadIdx.x >> 5);   // 0..1023
    const int b     = wid >> 8;
    const int ibase = (wid & 255) << 4;

    // Loop-invariant B tiles: B[kd][i] = K[ibase+i][kd]; lane ln reads K row ln,
    // d = kstep*32 + {8*half+e, 16+8*half+e}.
    ABReg Bk0, Bk1;
    {
        const unsigned short* krow = kb + (size_t)(b * SEQ + ibase + ln) * DH;
        const uint4* p0 = (const uint4*)(krow + 8 * half);
        const uint4* p1 = (const uint4*)(krow + 32 + 8 * half);
        Bk0.q[0] = p0[0]; Bk0.q[1] = p0[2];
        Bk1.q[0] = p1[0]; Bk1.q[1] = p1[2];
    }

    const float*          brow = bias + b * SEQ;
    const unsigned short* qbb  = qb + (size_t)(b * SEQ) * DH;
    const unsigned short* vtb  = vt + (size_t)(b * DH) * SEQ;

    float m_i = -3.0e38f, l_i = 0.0f;
    v8f o[4];
    #pragma unroll
    for (int n = 0; n < 4; ++n)
        #pragma unroll
        for (int r = 0; r < 8; ++r) o[n][r] = 0.0f;

    #pragma unroll 1
    for (int jb = 0; jb < SEQ; jb += 32) {
        // ================= issue ALL independent loads up-front =============
        // Q A-tiles: row j = jb + jt*16 + ln; elems: d = ks*32 + {8h+e, 16+8h+e}
        ABReg aq00, aq01, aq10, aq11;
        {
            const unsigned short* q0 = qbb + (size_t)(jb + ln) * DH;
            const unsigned short* q1 = qbb + (size_t)(jb + 16 + ln) * DH;
            aq00.q[0] = *(const uint4*)(q0 + 8 * half);
            aq00.q[1] = *(const uint4*)(q0 + 16 + 8 * half);
            aq01.q[0] = *(const uint4*)(q0 + 32 + 8 * half);
            aq01.q[1] = *(const uint4*)(q0 + 48 + 8 * half);
            aq10.q[0] = *(const uint4*)(q1 + 8 * half);
            aq10.q[1] = *(const uint4*)(q1 + 16 + 8 * half);
            aq11.q[0] = *(const uint4*)(q1 + 32 + 8 * half);
            aq11.q[1] = *(const uint4*)(q1 + 48 + 8 * half);
        }
        // V^T A-tiles for PV (needed only after softmax -> deep latency hiding)
        ABReg av[4];
        #pragma unroll
        for (int nt = 0; nt < 4; ++nt) {
            const unsigned short* vr = vtb + (size_t)(nt * 16 + ln) * SEQ
                                       + jb + 16 * half;
            av[nt].q[0] = *(const uint4*)(vr);
            av[nt].q[1] = *(const uint4*)(vr + 8);
        }
        // mask bias; elem r of St0 -> j = jb + r + 8*half, St1 -> +16
        float4 b0a = *(const float4*)(brow + jb + 8 * half);
        float4 b0b = *(const float4*)(brow + jb + 8 * half + 4);
        float4 b1a = *(const float4*)(brow + jb + 16 + 8 * half);
        float4 b1b = *(const float4*)(brow + jb + 16 + 8 * half + 4);
        // prefetch next block's Q / V rows (global_prefetch_b8)
        if (jb + 32 < SEQ) {
            __builtin_prefetch(qbb + (size_t)(jb + 32 + ln) * DH, 0, 3);
            __builtin_prefetch(vtb + (size_t)ln * SEQ + jb + 32, 0, 3);
        }

        // ================= score WMMAs =====================================
        v8f s0, s1;
        #pragma unroll
        for (int r = 0; r < 8; ++r) { s0[r] = 0.0f; s1[r] = 0.0f; }
        s0 = wmma_bf16(aq00.v, Bk0.v, s0);
        s0 = wmma_bf16(aq01.v, Bk1.v, s0);
        s1 = wmma_bf16(aq10.v, Bk0.v, s1);
        s1 = wmma_bf16(aq11.v, Bk1.v, s1);

        float sv0[8] = { s0[0] + b0a.x, s0[1] + b0a.y, s0[2] + b0a.z, s0[3] + b0a.w,
                         s0[4] + b0b.x, s0[5] + b0b.y, s0[6] + b0b.z, s0[7] + b0b.w };
        float sv1[8] = { s1[0] + b1a.x, s1[1] + b1a.y, s1[2] + b1a.z, s1[3] + b1a.w,
                         s1[4] + b1b.x, s1[5] + b1b.y, s1[6] + b1b.z, s1[7] + b1b.w };

        // ================= online softmax over j ===========================
        float mx = sv0[0];
        #pragma unroll
        for (int r = 1; r < 8; ++r) mx = fmaxf(mx, sv0[r]);
        #pragma unroll
        for (int r = 0; r < 8; ++r) mx = fmaxf(mx, sv1[r]);
        mx = fmaxf(mx, __shfl_xor(mx, 16, 32));
        float mnew  = fmaxf(m_i, mx);
        float alpha = __builtin_amdgcn_exp2f((m_i - mnew) * LOG2E);
        float rs = 0.0f;
        float p0[8], p1[8];
        #pragma unroll
        for (int r = 0; r < 8; ++r) {
            p0[r] = __builtin_amdgcn_exp2f((sv0[r] - mnew) * LOG2E); rs += p0[r];
            p1[r] = __builtin_amdgcn_exp2f((sv1[r] - mnew) * LOG2E); rs += p1[r];
        }
        rs += __shfl_xor(rs, 16, 32);
        l_i = l_i * alpha + rs;
        m_i = mnew;
        #pragma unroll
        for (int n = 0; n < 4; ++n)
            #pragma unroll
            for (int r = 0; r < 8; ++r) o[n][r] *= alpha;

        // ---- build P^T as B operand (32x16): lanes 0-15 carry k=0..15 (St0),
        // lanes 16-31 carry k=16..31 (St1); half-swap via shfl_xor(16).
        unsigned int pk0[4], pk1[4];
        #pragma unroll
        for (int r = 0; r < 4; ++r) {
            pk0[r] = pack2bf(p0[2 * r], p0[2 * r + 1]);
            pk1[r] = pack2bf(p1[2 * r], p1[2 * r + 1]);
        }
        ABReg bp;
        #pragma unroll
        for (int r = 0; r < 4; ++r) {
            unsigned int x0 = __shfl_xor(pk0[r], 16, 32);
            unsigned int x1 = __shfl_xor(pk1[r], 16, 32);
            bp.u[r]     = half ? x1     : pk0[r];
            bp.u[4 + r] = half ? pk1[r] : x0;
        }

        // ================= O^T += V^T * P^T ================================
        #pragma unroll
        for (int nt = 0; nt < 4; ++nt)
            o[nt] = wmma_bf16(av[nt].v, bp.v, o[nt]);
    }

    // O^T C-layout: elem r -> d = nt*16 + r + 8*half, lane -> row i = ibase + ln
    float inv = 1.0f / l_i;
    float* orow = out + (size_t)(b * SEQ + ibase + ln) * DH;
    #pragma unroll
    for (int nt = 0; nt < 4; ++nt)
        #pragma unroll
        for (int r = 0; r < 8; ++r)
            orow[nt * 16 + r + 8 * half] = o[nt][r] * inv;
}

// ---------------------------------------------------------------- launch ---
extern "C" void kernel_launch(void* const* d_in, const int* in_sizes, int n_in,
                              void* d_out, int out_size, void* d_ws, size_t ws_size,
                              hipStream_t stream) {
    const float* x   = (const float*)d_in[0];
    const int*   msk = (const int*)d_in[1];
    const float* Wk  = (const float*)d_in[2];
    const float* Wq  = (const float*)d_in[3];
    const float* Wv  = (const float*)d_in[4];
    float* out = (float*)d_out;

    char* ws = (char*)d_ws;
    const size_t mat_bytes = (size_t)BATCH * SEQ * DH * 2;   // 2 MB each
    unsigned short* kb   = (unsigned short*)(ws);
    unsigned short* qb   = (unsigned short*)(ws + mat_bytes);
    unsigned short* vt   = (unsigned short*)(ws + 2 * mat_bytes);
    float*          bias = (float*)(ws + 3 * mat_bytes);
    unsigned int*   wb   = (unsigned int*)(ws + 3 * mat_bytes
                                           + (size_t)BATCH * SEQ * 4);

    bias_kernel<<<(BATCH * SEQ) / 256, 256, 0, stream>>>(msk, bias);
    wprep_kernel<<<(3 * DH * HID / 2 + 255) / 256, 256, 0, stream>>>(Wk, Wq, Wv, wb);
    // 4 batches * 256 row-tiles = 1024 waves (each does K+Q+V), 4 waves/block
    proj_kernel<<<256, 128, 0, stream>>>(x, (const unsigned short*)wb, kb, qb, vt);
    // 4 batches * 256 row-tiles = 1024 waves, 4 waves/block
    attn_kernel<<<256, 128, 0, stream>>>(kb, qb, vt, bias, out);
}
